// ALiBiMultiheadAttention_47949014893018
// MI455X (gfx1250) — matrix operand order repaired and checked by reading, not compiled, not run on hardware
//
#include <hip/hip_runtime.h>
#include <hip/hip_bf16.h>
#include <math.h>

// ---------------------------------------------------------------------------
// ALiBi multi-head attention for MI455X (gfx1250).
// bf16 WMMA everywhere (v_wmma_f32_16x16x32_bf16), flash softmax, TDM-staged
// weight tiles. B=4, S=1024, D=1024, H=16, DK=64, NPAD via mask input.
// ---------------------------------------------------------------------------

typedef __bf16 bf16_t;
typedef __attribute__((ext_vector_type(16))) __bf16 v16bf;
typedef __attribute__((ext_vector_type(8)))  __bf16 v8bf;
typedef __attribute__((ext_vector_type(8)))  float  v8f;
typedef __attribute__((ext_vector_type(4))) unsigned int u32x4;
typedef __attribute__((ext_vector_type(8))) int i32x8;
typedef __attribute__((ext_vector_type(4))) int i32x4;

#define NEG_BIG (-1.0e30f)

#if defined(__has_builtin)
#if __has_builtin(__builtin_amdgcn_tensor_load_to_lds) && \
    __has_builtin(__builtin_amdgcn_s_wait_tensorcnt)
#define HAVE_TDM 1
#endif
#endif

static __device__ inline v16bf pack16(const bf16_t* lo, const bf16_t* hi) {
  v8bf a = *(const v8bf*)lo;
  v8bf b = *(const v8bf*)hi;
  v16bf r;
#pragma unroll
  for (int e = 0; e < 8; ++e) { r[e] = a[e]; r[e + 8] = b[e]; }
  return r;
}

static __device__ inline v8f wmma_bf16(v16bf a, v16bf b, v8f c) {
  return __builtin_amdgcn_wmma_f32_16x16x32_bf16(false, a, false, b, (short)0, c,
                                                 false, false);
}

#ifdef HAVE_TDM
// Issue a TDM load of a 2-D bf16 tile (tile_dim0=32 elems along k, tile_dim1=64
// rows, row stride = 1024 elems) from global `gaddr` into LDS offset `ldsoff`.
// D# layout per CDNA5 ISA ch.8 (group0: count/lds/global/type, group1: dims).
static __device__ inline void tdm_load_w_tile(unsigned long long gaddr,
                                              unsigned int ldsoff) {
  u32x4 g0;
  g0[0] = 1u;                                    // count=1, user descriptor
  g0[1] = ldsoff;                                // lds_addr (bytes)
  g0[2] = (unsigned int)gaddr;                   // global_addr[31:0]
  g0[3] = (unsigned int)((gaddr >> 32) & 0x01FFFFFFu) | (2u << 30);  // type=2
  i32x8 g1;
  g1[0] = 0x00010000;                 // wg_mask=0, data_size=1 (2 bytes)
  g1[1] = (int)0xFFFF0000u;           // tensor_dim0 lo16 (0x7FFFFFFF: no OOB)
  g1[2] = (int)0xFFFF7FFFu;           // tensor_dim0 hi16 | tensor_dim1 lo16
  g1[3] = 0x7FFF | (32 << 16);        // tensor_dim1 hi16 | tile_dim0=32
  g1[4] = 64;                         // tile_dim1=64, tile_dim2=0
  g1[5] = 1024;                       // tensor_dim0_stride = 1024 elems
  g1[6] = 0;
  g1[7] = 0;
  i32x4 gz = (i32x4)(0);
#if __clang_major__ >= 23
  __builtin_amdgcn_tensor_load_to_lds(g0, g1, gz, gz, (i32x8)(0), 0);
#else
  __builtin_amdgcn_tensor_load_to_lds(g0, g1, gz, gz, 0);
#endif
}
#endif

// ---------------------------------------------------------------------------
// fp32 -> bf16 conversion (4 elements / thread)
// ---------------------------------------------------------------------------
__global__ void cvt_f32_to_bf16(const float* __restrict__ in,
                                bf16_t* __restrict__ out, int n4) {
  int i = blockIdx.x * blockDim.x + threadIdx.x;
  if (i < n4) {
    float4 v = ((const float4*)in)[i];
    bf16_t* o = out + (size_t)i * 4;
    o[0] = (bf16_t)v.x; o[1] = (bf16_t)v.y;
    o[2] = (bf16_t)v.z; o[3] = (bf16_t)v.w;
  }
}

// ---------------------------------------------------------------------------
// Y[m,n] = sum_k X[m,k] * W[n,k] + bias[n]     (M=4096, N=K=1024, bf16 WMMA)
// Block = 4 waves sharing one 64-col strip; the 64x32 weight tile is staged
// into LDS by the Tensor Data Mover (double-buffered, overlapped with math).
// MODE 0: bf16 [B,H,S,DK] ; MODE 1: bf16 [B,H,DK,S] ; MODE 2: f32 [M,N]
// ---------------------------------------------------------------------------
template <int MODE>
__global__ __launch_bounds__(128) void proj_gemm(const bf16_t* __restrict__ X,
                                                 const bf16_t* __restrict__ W,
                                                 const float* __restrict__ bias,
                                                 void* __restrict__ out) {
  constexpr int K = 1024, N = 1024;
  __shared__ __align__(16) bf16_t Wlds[2][64 * 32];  // two 4KB tile buffers

  const int lane = threadIdx.x & 31;
  const int wave = threadIdx.x >> 5;
  const int half = lane >> 4;
  const int l16  = lane & 15;
  const int m0 = (blockIdx.x * 4 + wave) * 32;
  const int n0 = blockIdx.y * 64;

  v8f acc[2][4];
#pragma unroll
  for (int a = 0; a < 2; ++a)
#pragma unroll
    for (int c = 0; c < 4; ++c) acc[a][c] = (v8f)(0.0f);

  const bf16_t* A0 = X + (size_t)(m0 + l16) * K;
  const bf16_t* A1 = X + (size_t)(m0 + 16 + l16) * K;

#ifdef HAVE_TDM
  const unsigned long long wbase =
      (unsigned long long)(uintptr_t)(W + (size_t)n0 * K);
  const unsigned int lds0 = (unsigned int)(uintptr_t)&Wlds[0][0];
  const unsigned int lds1 = (unsigned int)(uintptr_t)&Wlds[1][0];
  if (wave == 0) tdm_load_w_tile(wbase, lds0);
#endif

  for (int k = 0; k < K; k += 32) {
    const int buf = (k >> 5) & 1;
#ifdef HAVE_TDM
    if (wave == 0) {
      if (k + 32 < K) {  // prefetch next tile into other buffer, keep 1 inflight
        tdm_load_w_tile(wbase + (unsigned long long)(k + 32) * 2,
                        buf ? lds0 : lds1);
        __builtin_amdgcn_s_wait_tensorcnt((short)1);
      } else {
        __builtin_amdgcn_s_wait_tensorcnt((short)0);
      }
    }
    __syncthreads();   // publish TDM-written tile to all waves
#else
    __syncthreads();   // previous iteration finished reading Wlds
    {                  // cooperative copy: 128 threads x 16 bf16 = 64x32 tile
      const int tid = threadIdx.x;
      const bf16_t* src = W + (size_t)(n0 + (tid >> 1)) * K + k + (tid & 1) * 16;
      bf16_t* dst = &Wlds[buf][(tid >> 1) * 32 + (tid & 1) * 16];
      *(v8bf*)dst       = *(const v8bf*)src;
      *(v8bf*)(dst + 8) = *(const v8bf*)(src + 8);
    }
    __syncthreads();
#endif

    // A frag (16x32 bf16): lane row = l16, K = half*8+e (lo) / 16+half*8+e (hi)
    v16bf a0 = pack16(A0 + k + half * 8, A0 + k + half * 8 + 16);
    v16bf a1 = pack16(A1 + k + half * 8, A1 + k + half * 8 + 16);
#pragma unroll
    for (int nt = 0; nt < 4; ++nt) {
      // B frag (32x16) from LDS: col = l16 within tile row, K = half*8+e (lo) / 16+half*8+e (hi)
      const bf16_t* wp = &Wlds[buf][(nt * 16 + l16) * 32 + half * 8];
      v16bf bm = pack16(wp, wp + 16);
      acc[0][nt] = wmma_bf16(a0, bm, acc[0][nt]);
      acc[1][nt] = wmma_bf16(a1, bm, acc[1][nt]);
    }
#ifdef HAVE_TDM
    __syncthreads();   // all waves done with Wlds[buf] before TDM overwrites it
#endif
  }

#pragma unroll
  for (int nt = 0; nt < 4; ++nt) {
    const int n = n0 + nt * 16 + l16;
    const float bv = bias[n];
#pragma unroll
    for (int mt = 0; mt < 2; ++mt) {
#pragma unroll
      for (int i = 0; i < 8; ++i) {
        const int m = m0 + mt * 16 + half * 8 + i;   // C layout: row = half*8+i
        const float v = acc[mt][nt][i] + bv;
        if (MODE == 2) {
          ((float*)out)[(size_t)m * N + n] = v;
        } else {
          const int b = m >> 10, s = m & 1023;
          const int h = n >> 6, d = n & 63;
          const size_t idx =
              (MODE == 0) ? (((size_t)(b * 16 + h) * 1024 + s) * 64 + d)
                          : (((size_t)(b * 16 + h) * 64 + d) * 1024 + s);
          ((bf16_t*)out)[idx] = (bf16_t)v;
        }
      }
    }
  }
}

// ---------------------------------------------------------------------------
// Flash attention with ALiBi + causal + key-padding mask.
// One wave per 16-query tile of one (b,h); 64-key blocks (16 WMMAs / block).
// Qp,Kp: [B,H,S,64] bf16 ; Vt: [B,H,64,S] bf16 (transposed) ; O: [B,S,D] bf16
// ---------------------------------------------------------------------------
__global__ __launch_bounds__(128) void flash_alibi(
    const bf16_t* __restrict__ Qp, const bf16_t* __restrict__ Kp,
    const bf16_t* __restrict__ Vt, const unsigned char* __restrict__ padm,
    bf16_t* __restrict__ O) {
  constexpr int S = 1024, DK = 64, D = 1024;
  __shared__ __align__(16) bf16_t Pbuf[4][16 * 64];  // per-wave P tile (16x64)

  const int lane = threadIdx.x & 31;
  const int wave = threadIdx.x >> 5;
  const int half = lane >> 4;
  const int l16  = lane & 15;
  const int q0 = (blockIdx.x * 4 + wave) * 16;
  const int bh = blockIdx.y;          // b*H + h
  const int b = bh >> 4, h = bh & 15;

  const bf16_t* Qb = Qp + (size_t)bh * S * DK;
  const bf16_t* Kb = Kp + (size_t)bh * S * DK;
  const bf16_t* Vb = Vt + (size_t)bh * DK * S;
  const float slope = exp2f(-0.5f * (float)(h + 1));  // 2^(-8*(h+1)/16)

  // Preload Q A-fragments for the two 32-wide dk chunks (reused every block).
  v16bf qa[2];
#pragma unroll
  for (int c = 0; c < 2; ++c) {
    const bf16_t* p = Qb + (size_t)(q0 + l16) * DK + c * 32 + half * 8;
    qa[c] = pack16(p, p + 16);
  }

  v8f o[4];
#pragma unroll
  for (int t = 0; t < 4; ++t) o[t] = (v8f)(0.0f);
  float mrow[8], lrow[8];
#pragma unroll
  for (int i = 0; i < 8; ++i) { mrow[i] = NEG_BIG; lrow[i] = 0.0f; }

  const int kmax = (q0 + 16 < S) ? q0 + 16 : S;   // causal upper bound
  for (int kb = 0; kb < kmax; kb += 64) {
    // ---- scores S = Q K^T : 16q x 64keys in four 16-key accumulators ----
    v8f sc[4];
#pragma unroll
    for (int t = 0; t < 4; ++t) sc[t] = (v8f)(0.0f);
#pragma unroll
    for (int c = 0; c < 2; ++c)
#pragma unroll
      for (int t = 0; t < 4; ++t) {
        const bf16_t* p =
            Kb + (size_t)(kb + t * 16 + l16) * DK + c * 32 + half * 8;
        sc[t] = wmma_bf16(qa[c], pack16(p, p + 16), sc[t]);
      }

    // Lane-precomputed ALiBi/mask terms (bias(i) = slope*dt - slope*i; masked
    // iff dt > i, padding forces the threshold above any i).
    int dm[4];
    float fb[4];
#pragma unroll
    for (int t = 0; t < 4; ++t) {
      const int kt = kb + t * 16 + l16;
      const int dt = kt - q0 - half * 8;
      fb[t] = slope * (float)dt;
      dm[t] = (padm[b * S + kt] != 0) ? 1000 : dt;
    }

    // ---- scale + bias + mask + online softmax (rows live in lane halves) ---
#pragma unroll
    for (int i = 0; i < 8; ++i) {
      const float si = slope * (float)i;
      float v[4];
#pragma unroll
      for (int t = 0; t < 4; ++t) {
        const float x = fmaf(sc[t][i], 0.125f, fb[t] - si);
        v[t] = (dm[t] > i) ? NEG_BIG : x;
      }
      float tmax = fmaxf(fmaxf(v[0], v[1]), fmaxf(v[2], v[3]));
      tmax = fmaxf(tmax, __shfl_xor(tmax, 1));
      tmax = fmaxf(tmax, __shfl_xor(tmax, 2));
      tmax = fmaxf(tmax, __shfl_xor(tmax, 4));
      tmax = fmaxf(tmax, __shfl_xor(tmax, 8));
      const float mn = fmaxf(mrow[i], tmax);
      const float rescale = __expf(mrow[i] - mn);
      mrow[i] = mn;

      float p[4];
      float r = 0.0f;
#pragma unroll
      for (int t = 0; t < 4; ++t) { p[t] = __expf(v[t] - mn); r += p[t]; }
      r += __shfl_xor(r, 1);
      r += __shfl_xor(r, 2);
      r += __shfl_xor(r, 4);
      r += __shfl_xor(r, 8);
      lrow[i] = lrow[i] * rescale + r;

#pragma unroll
      for (int t4 = 0; t4 < 4; ++t4) o[t4][i] *= rescale;

      // P row-major [16q][64k] into wave-private LDS (C layout -> memory)
#pragma unroll
      for (int t = 0; t < 4; ++t)
        Pbuf[wave][(half * 8 + i) * 64 + t * 16 + l16] = (bf16_t)p[t];
    }

    // Wave-private LDS store->load turnaround (DS in-order; explicit wait).
    asm volatile("s_wait_dscnt 0" ::: "memory");

    // Reload P in A-fragment layout for the two 32-key K-chunks.
    v16bf pa[2];
#pragma unroll
    for (int c = 0; c < 2; ++c) {
      const bf16_t* pp = &Pbuf[wave][l16 * 64 + c * 32 + half * 8];
      pa[c] = pack16(pp, pp + 16);
    }

    // ---- O += P V : B frag col = dv (l16), K = key (contiguous in Vt) ----
#pragma unroll
    for (int c = 0; c < 2; ++c)
#pragma unroll
      for (int t4 = 0; t4 < 4; ++t4) {
        const bf16_t* vp =
            Vb + (size_t)(t4 * 16 + l16) * S + kb + c * 32 + half * 8;
        o[t4] = wmma_bf16(pa[c], pack16(vp, vp + 16), o[t4]);
      }
  }

#pragma unroll
  for (int i = 0; i < 8; ++i) lrow[i] = 1.0f / lrow[i];
#pragma unroll
  for (int t4 = 0; t4 < 4; ++t4) {
#pragma unroll
    for (int i = 0; i < 8; ++i) {
      const int qpos = q0 + half * 8 + i;
      O[((size_t)(b * S + qpos)) * D + h * DK + t4 * 16 + l16] =
          (bf16_t)(o[t4][i] * lrow[i]);
    }
  }
}

// ---------------------------------------------------------------------------
extern "C" void kernel_launch(void* const* d_in, const int* in_sizes, int n_in,
                              void* d_out, int out_size, void* d_ws,
                              size_t ws_size, hipStream_t stream) {
  (void)in_sizes; (void)n_in; (void)out_size; (void)ws_size;

  const float* query = (const float*)d_in[0];
  const float* key   = (const float*)d_in[1];
  const float* value = (const float*)d_in[2];
  // d_in[3] (alibi_bias, 64MB) is reproduced analytically: slope_h*(k-q).
  const unsigned char* padm = (const unsigned char*)d_in[4];
  const float* Wq = (const float*)d_in[5];
  const float* bq = (const float*)d_in[6];
  const float* Wk = (const float*)d_in[7];
  const float* bk = (const float*)d_in[8];
  const float* Wv = (const float*)d_in[9];
  const float* bv = (const float*)d_in[10];
  const float* Wo = (const float*)d_in[11];
  const float* bo = (const float*)d_in[12];
  float* out = (float*)d_out;

  constexpr size_t XE = 4ull * 1024 * 1024;  // B*S*D elements
  constexpr size_t WE = 1024ull * 1024;      // D*D elements

  bf16_t* qX  = (bf16_t*)d_ws;     // activations / weights in bf16 (64MB total)
  bf16_t* kX  = qX + XE;
  bf16_t* vX  = kX + XE;
  bf16_t* Wqb = vX + XE;
  bf16_t* Wkb = Wqb + WE;
  bf16_t* Wvb = Wkb + WE;
  bf16_t* Wob = Wvb + WE;
  bf16_t* Qp  = Wob + WE;          // [B,H,S,DK]
  bf16_t* Kp  = Qp + XE;           // [B,H,S,DK]
  bf16_t* Vt  = Kp + XE;           // [B,H,DK,S]
  bf16_t* AO  = Vt + XE;           // attention output [B,S,D]

  const int xb = (int)(XE / 4), wb = (int)(WE / 4);
  cvt_f32_to_bf16<<<dim3((xb + 255) / 256), dim3(256), 0, stream>>>(query, qX, xb);
  cvt_f32_to_bf16<<<dim3((xb + 255) / 256), dim3(256), 0, stream>>>(key, kX, xb);
  cvt_f32_to_bf16<<<dim3((xb + 255) / 256), dim3(256), 0, stream>>>(value, vX, xb);
  cvt_f32_to_bf16<<<dim3((wb + 255) / 256), dim3(256), 0, stream>>>(Wq, Wqb, wb);
  cvt_f32_to_bf16<<<dim3((wb + 255) / 256), dim3(256), 0, stream>>>(Wk, Wkb, wb);
  cvt_f32_to_bf16<<<dim3((wb + 255) / 256), dim3(256), 0, stream>>>(Wv, Wvb, wb);
  cvt_f32_to_bf16<<<dim3((wb + 255) / 256), dim3(256), 0, stream>>>(Wo, Wob, wb);

  dim3 gg(32, 16), gb(128);  // 4096x1024 output, 32x64 per wave, 4 waves/block
  proj_gemm<0><<<gg, gb, 0, stream>>>(qX, Wqb, bq, (void*)Qp);
  proj_gemm<0><<<gg, gb, 0, stream>>>(kX, Wkb, bk, (void*)Kp);
  proj_gemm<1><<<gg, gb, 0, stream>>>(vX, Wvb, bv, (void*)Vt);

  flash_alibi<<<dim3(16, 64), dim3(128), 0, stream>>>(Qp, Kp, Vt, padm, AO);

  proj_gemm<2><<<gg, gb, 0, stream>>>(AO, Wob, bo, (void*)out);
}
